// SDPACoreAttention_37787122270761
// MI455X (gfx1250) — matrix operand order repaired and checked by reading, not compiled, not run on hardware
//
#include <hip/hip_runtime.h>
#include <hip/hip_bf16.h>

// ---------------------------------------------------------------------------
// Varlen packed GQA flash attention for gfx1250 (MI455X), wave32 + WMMA.
//   T=4096 tokens, NQ=16 q-heads, NK=4 kv-heads, HN=128 head dim.
//   Workgroup = 4 waves = the 4 query heads sharing one KV head, one 16-row
//   q-tile. K/V tiles are staged to LDS ONCE per workgroup (4x less L2/HBM
//   traffic than per-wave staging under GQA). Each wave runs its own online
//   softmax and accumulates O via v_wmma_f32_16x16x32_f16 (f32 accum).
// ---------------------------------------------------------------------------

typedef __attribute__((ext_vector_type(16))) _Float16 v16h;
typedef __attribute__((ext_vector_type(8)))  _Float16 v8h;
typedef __attribute__((ext_vector_type(4)))  _Float16 v4h;
typedef __attribute__((ext_vector_type(8)))  float    v8f;

#define T_TOK 4096
#define NQH   16
#define NKH   4
#define GQA   (NQH / NKH)   // 4 q-heads per kv head
#define HN    128
#define QT    16            // query rows per wave
#define KT    32            // keys per inner iteration (== WMMA K for PV)

__launch_bounds__(32 * GQA)
__global__ void sdpa_varlen_wmma_kernel(const float* __restrict__ Q,
                                        const float* __restrict__ K,
                                        const float* __restrict__ V,
                                        const int*   __restrict__ cu,
                                        float*       __restrict__ O) {
    __shared__ _Float16 Ksh[KT][HN];        // K tile, row-major            (8 KB)
    __shared__ _Float16 VshT[HN][KT];       // V tile, TRANSPOSED [d][k]    (8 KB)
    __shared__ _Float16 Psh[GQA][QT][KT];   // per-wave probabilities       (4 KB)

    const int tid  = threadIdx.x;          // 0..127
    const int w    = tid >> 5;             // wave id == q-head within group
    const int lane = tid & 31;             // 0..31 (wave32)
    const int half = lane >> 4;            // lane group: 0 or 1
    const int ln   = lane & 15;            // lane within group
    const int hk   = blockIdx.y;           // kv head
    const int h    = hk * GQA + w;         // query head
    const int qbase = blockIdx.x * QT;

    // ---- segment lookup (block-diagonal varlen mask) ----
    int seg_end = T_TOK, seg_start = 0;
    #pragma unroll
    for (int s = 0; s < 4; ++s) {
        if (qbase >= cu[s] && qbase < cu[s + 1]) {
            seg_start = cu[s];
            seg_end   = cu[s + 1];
        }
    }

    // ---- load Q tile into WMMA A-layout registers (f32 -> f16) ----
    // A 16x32 f16, contiguous k order chosen here: lane m = ln, element e -> K = 16*half + e
    v16h Aq[4];
    {
        const float* qp = Q + ((size_t)(qbase + ln) * NQH + h) * HN;
        #pragma unroll
        for (int c = 0; c < 4; ++c) {
            const int d0 = 32 * c + 16 * half;
            #pragma unroll
            for (int e = 0; e < 8; ++e) Aq[c][e]     = (_Float16)qp[d0 + e];
            #pragma unroll
            for (int e = 0; e < 8; ++e) Aq[c][8 + e] = (_Float16)qp[d0 + 8 + e];
        }
    }

    // ---- accumulators: 8 x (16x16 f32) output tiles + online softmax state ----
    v8f acc[8];
    #pragma unroll
    for (int t = 0; t < 8; ++t) acc[t] = v8f{};
    float mrow[8], lrow[8];
    #pragma unroll
    for (int r = 0; r < 8; ++r) { mrow[r] = -INFINITY; lrow[r] = 0.0f; }

    const float scale = 0.08838834764831845f;      // 128^-0.5
    const int   kmax  = min(seg_end, qbase + QT);  // exclusive causal bound
    // cooperative-load coordinates: each of 128 threads owns a quarter row
    const int ldrow = tid >> 2;          // 0..31
    const int lddp  = (tid & 3) * 32;    // d-part: 0,32,64,96

    for (int kb = seg_start; kb < kmax; kb += KT) {
        // ---- stage K tile (row-major f16) and V tile (transposed f16) in LDS.
        //      Shared by all 4 q-heads in this workgroup: loaded ONCE. ----
        {
            const int  krow = min(kb + ldrow, T_TOK - 1);   // clamp; masked later
            const float* kp = K + ((size_t)krow * NKH + hk) * HN + lddp;
            const float* vp = V + ((size_t)krow * NKH + hk) * HN + lddp;
            #pragma unroll
            for (int d = 0; d < 32; d += 4) {
                float4 kf = *(const float4*)(kp + d);
                *(v4h*)&Ksh[ldrow][lddp + d] =
                    v4h{(_Float16)kf.x, (_Float16)kf.y, (_Float16)kf.z, (_Float16)kf.w};
                float4 vf = *(const float4*)(vp + d);
                VshT[lddp + d + 0][ldrow] = (_Float16)vf.x;
                VshT[lddp + d + 1][ldrow] = (_Float16)vf.y;
                VshT[lddp + d + 2][ldrow] = (_Float16)vf.z;
                VshT[lddp + d + 3][ldrow] = (_Float16)vf.w;
            }
            // prefetch next tile into cache while we compute
            if (kb + KT < kmax) {
                const int nrow = min(kb + KT + ldrow, T_TOK - 1);
                __builtin_prefetch(K + ((size_t)nrow * NKH + hk) * HN + lddp, 0, 1);
                __builtin_prefetch(V + ((size_t)nrow * NKH + hk) * HN + lddp, 0, 1);
            }
        }
        __syncthreads();

        // ---- scores S = Q K^T : two 16x16 tiles (keys 0-15 / 16-31) ----
        v8f S0 = v8f{}, S1 = v8f{};
        #pragma unroll
        for (int c = 0; c < 4; ++c) {
            // B 32x16, same contiguous k order as Aq: lane n = ln, element e -> K(d) = c*32 + 16*half + e
            v16h B0 = *(const v16h*)&Ksh[ln][c * 32 + 16 * half];
            v16h B1 = *(const v16h*)&Ksh[16 + ln][c * 32 + 16 * half];
            S0 = __builtin_amdgcn_wmma_f32_16x16x32_f16(false, Aq[c], false, B0,
                                                        (short)0, S0, false, false);
            S1 = __builtin_amdgcn_wmma_f32_16x16x32_f16(false, Aq[c], false, B1,
                                                        (short)0, S1, false, false);
        }

        // ---- mask + online softmax (C layout: lane col N=ln, reg r row M=r+8*half)
        #pragma unroll
        for (int r = 0; r < 8; ++r) {
            const int qg  = qbase + r + 8 * half;
            const int kg0 = kb + ln;
            const int kg1 = kb + 16 + ln;
            float s0 = (kg0 <= qg && kg0 < seg_end) ? S0[r] * scale : -INFINITY;
            float s1 = (kg1 <= qg && kg1 < seg_end) ? S1[r] * scale : -INFINITY;

            float mx = fmaxf(s0, s1);
            #pragma unroll
            for (int off = 1; off < 16; off <<= 1)
                mx = fmaxf(mx, __shfl_xor(mx, off, 32));
            const float mnew = fmaxf(mrow[r], mx);
            const float corr = __expf(mrow[r] - mnew);
            const float p0 = __expf(s0 - mnew);
            const float p1 = __expf(s1 - mnew);
            float rs = p0 + p1;
            #pragma unroll
            for (int off = 1; off < 16; off <<= 1)
                rs += __shfl_xor(rs, off, 32);
            lrow[r] = lrow[r] * corr + rs;
            mrow[r] = mnew;
            #pragma unroll
            for (int t = 0; t < 8; ++t) acc[t][r] *= corr;

            Psh[w][r + 8 * half][ln]      = (_Float16)p0;
            Psh[w][r + 8 * half][16 + ln] = (_Float16)p1;
        }
        __syncthreads();

        // ---- reshape P into A-layout and accumulate O += P * V ----
        v8h lo = *(const v8h*)&Psh[w][ln][16 * half];       // K = 16*half + e
        v8h hi = *(const v8h*)&Psh[w][ln][16 * half + 8];   // K = 16*half + 8 + e
        v16h Ap = __builtin_shufflevector(lo, hi, 0, 1, 2, 3, 4, 5, 6, 7,
                                          8, 9, 10, 11, 12, 13, 14, 15);
        #pragma unroll
        for (int t = 0; t < 8; ++t) {
            // B 32x16, same contiguous k order as Ap: lane n = d = t*16+ln, element e -> K(key) = 16*half + e
            v16h Bv = *(const v16h*)&VshT[t * 16 + ln][16 * half];
            acc[t] = __builtin_amdgcn_wmma_f32_16x16x32_f16(false, Ap, false, Bv,
                                                            (short)0, acc[t],
                                                            false, false);
        }
    }

    // ---- epilogue: normalize and store [tok][h*HN + d] (fp32) ----
    #pragma unroll
    for (int r = 0; r < 8; ++r) {
        const float inv = 1.0f / lrow[r];
        float* op = O + (size_t)(qbase + r + 8 * half) * (NQH * HN) + h * HN;
        #pragma unroll
        for (int t = 0; t < 8; ++t)
            op[t * 16 + ln] = acc[t][r] * inv;
    }
}

extern "C" void kernel_launch(void* const* d_in, const int* in_sizes, int n_in,
                              void* d_out, int out_size, void* d_ws, size_t ws_size,
                              hipStream_t stream) {
    (void)in_sizes; (void)n_in; (void)out_size; (void)d_ws; (void)ws_size;
    const float* q  = (const float*)d_in[0];
    const float* k  = (const float*)d_in[1];
    const float* v  = (const float*)d_in[2];
    const int*   cu = (const int*)d_in[3];
    float* out = (float*)d_out;

    dim3 grid(T_TOK / QT, NKH);    // 256 q-tiles x 4 kv heads
    dim3 block(32 * GQA);          // 4 waves = 4 q-heads sharing the KV tile
    sdpa_varlen_wmma_kernel<<<grid, block, 0, stream>>>(q, k, v, cu, out);
}
